// Lemmatizer_13460427506233
// MI455X (gfx1250) — matrix operand order repaired and checked by reading, not compiled, not run on hardware
//
#include <hip/hip_runtime.h>
#include <hip/hip_bf16.h>
#include <math.h>

// ---------------------------------------------------------------------------
// Types / helpers
// ---------------------------------------------------------------------------
typedef __attribute__((ext_vector_type(8)))  _Float16 v8h;
typedef __attribute__((ext_vector_type(16))) _Float16 v16h;
typedef __attribute__((ext_vector_type(8)))  float    v8f;

__device__ __forceinline__ float sigf(float x) { return 1.0f / (1.0f + __expf(-x)); }

// CDNA5 async global->LDS copy (16B per lane, ASYNCcnt-tracked).
// lds: generic pointer to __shared__ (low 32 bits = wave LDS offset)
__device__ __forceinline__ void async_ld128(void* lds, const void* gptr)
{
  unsigned l = (unsigned)(unsigned long long)lds;
  unsigned long long g = (unsigned long long)gptr;
  asm volatile("global_load_async_to_lds_b128 %0, %1, off" :: "v"(l), "v"(g) : "memory");
}
__device__ __forceinline__ void wait_async_le2()
{
  asm volatile("s_wait_asynccnt 0x2" ::: "memory");
}
__device__ __forceinline__ void wait_async_0()
{
  asm volatile("s_wait_asynccnt 0x0" ::: "memory");
}

// Model constants
#define KB   1024     // batch
#define KT   48       // encoder time
#define KTH  52       // encoder time with halo (2 + 48 + 2)
#define KS   48       // decoder steps
#define KVV  512      // vocab
#define KCE  100      // char emb
#define KCOND 112
#define KCS0 212
#define KCS0P 224
#define KCCH 512
#define KHENC 256
#define KENC_OUT 624
#define KENC_OUTP 640
#define KHDEC 512
#define KATT 128
#define KDEC_IN 724
#define KDEC_INP 736

// ---------------------------------------------------------------------------
// Generic WMMA GEMM:  C[M,N] = A[M,Kp] * B[N,Kp]^T  (+bias[n]) (+=C)
// A,B are f16 with Kp a multiple of 32.  C is f32 with row stride ldc.
// Optional affine A-row remap: row' = (m / rps) * sstr + (m % rps) + roff
// (used for conv taps over a zero-halo'd [B, 52, C] activation buffer).
// Block = 256 threads (8 waves); 64x64 output tile; each wave owns a 16x32
// output region (two 16x16 WMMA accumulators).
// K-loop is double-buffered with GLOBAL_LOAD_ASYNC_TO_LDS_B128: tile k+1's
// DMA is issued before tile k's WMMA, then s_wait_asynccnt 2 drains only the
// oldest (current-tile) pair, keeping the new pair in flight.
// ---------------------------------------------------------------------------
__global__ void __launch_bounds__(256)
gemm_kernel(const _Float16* __restrict__ A, const _Float16* __restrict__ B,
            float* __restrict__ C, const float* __restrict__ bias,
            int Kp, int lda, int ldc, int accumulate,
            int rps, int sstr, int roff)
{
  __shared__ _Float16 As[2][64][32];
  __shared__ _Float16 Bs[2][64][32];

  const int tid  = threadIdx.x;
  const int n0   = blockIdx.x * 64;
  const int m0   = blockIdx.y * 64;
  const int lr   = tid >> 2;         // 0..63 : tile row loaded by this thread
  const int cseg = (tid & 3) * 8;    // 0,8,16,24 : 8-half segment

  // A row mapping (fixed per thread across the whole K loop)
  const int am = m0 + lr;
  size_t arow;
  if (rps > 0) arow = (size_t)(am / rps) * (size_t)sstr + (size_t)(am % rps) + (size_t)roff;
  else         arow = (size_t)am;
  const _Float16* aptr = A + arow * (size_t)lda + cseg;
  const _Float16* bptr = B + (size_t)(n0 + lr) * (size_t)Kp + cseg;

  const int wave  = tid >> 5;
  const int lane  = tid & 31;
  const int mt    = wave & 3;            // 4 m-subtiles
  const int ntb   = (wave >> 2) * 2;     // 2 n-subtiles per wave
  const int lrow  = mt * 16 + (lane & 15);
  const int abase = (lane < 16) ? 0 : 16;  // A frag K base (contiguous, matches B)
  const int bbase = (lane < 16) ? 0 : 16;  // B frag K base (contiguous, matches A)

  v8f accs[2]; accs[0] = (v8f){}; accs[1] = (v8f){};

  const int nk = Kp >> 5;
  // prologue: DMA tile 0 into buffer 0
  async_ld128(&As[0][lr][cseg], aptr);
  async_ld128(&Bs[0][lr][cseg], bptr);

  for (int kk = 0; kk < nk; ++kk) {
    const int cur = kk & 1;
    if (kk + 1 < nk) {
      // issue next tile's DMA into the other buffer (last read one iter ago,
      // protected by the trailing barrier of the previous iteration)
      const int koff = (kk + 1) << 5;
      __builtin_prefetch(bptr + koff + 64, 0, 1);
      async_ld128(&As[cur ^ 1][lr][cseg], aptr + koff);
      async_ld128(&Bs[cur ^ 1][lr][cseg], bptr + koff);
      wait_async_le2();   // current tile's pair has landed; next pair in flight
    } else {
      wait_async_0();
    }
    __syncthreads();

    // A fragment: lanes<16 hold K {0..15}, lanes>=16 hold K {16..31}
    v8h alo = *(const v8h*)&As[cur][lrow][abase];
    v8h ahi = *(const v8h*)&As[cur][lrow][abase + 8];
    v16h af = __builtin_shufflevector(alo, ahi, 0,1,2,3,4,5,6,7,8,9,10,11,12,13,14,15);

    // B fragments: contiguous 16 K-halves per lane-column
    v16h bf0 = *(const v16h*)&Bs[cur][(ntb + 0) * 16 + (lane & 15)][bbase];
    v16h bf1 = *(const v16h*)&Bs[cur][(ntb + 1) * 16 + (lane & 15)][bbase];

    accs[0] = __builtin_amdgcn_wmma_f32_16x16x32_f16(false, af, false, bf0,
                                                     (short)0, accs[0], false, false);
    accs[1] = __builtin_amdgcn_wmma_f32_16x16x32_f16(false, af, false, bf1,
                                                     (short)0, accs[1], false, false);
    __syncthreads();
  }

  // Epilogue per ISA C layout: VGPR r -> M = r (+8 for hi lanes), N = lane%16
  const int rowb = m0 + mt * 16 + ((lane < 16) ? 0 : 8);
  for (int i = 0; i < 2; ++i) {
    const int col = n0 + (ntb + i) * 16 + (lane & 15);
    const float bv = bias ? bias[col] : 0.0f;
    for (int r = 0; r < 8; ++r) {
      size_t off = (size_t)(rowb + r) * (size_t)ldc + (size_t)col;
      float v = accs[i][r] + bv;
      if (accumulate) v += C[off];
      C[off] = v;
    }
  }
}

// ---------------------------------------------------------------------------
// Encoder LSTM scan: one workgroup handles 16 batch rows across all T steps.
// gates[16,1024] = xw[b,t,:] + h[16,256] @ Whh[1024,256]^T   (WMMA)
// then the LSTM cell update; h kept in LDS as f16, c as f32.
// Whh fragments are direct 32B v16h loads from the L2-resident f16 weights.
// Dynamic LDS: h 8KB + c 16KB + gates 64KB = 88KB.
// ---------------------------------------------------------------------------
__global__ void __launch_bounds__(256)
enc_scan_kernel(const float* __restrict__ xw,      // [B*T, 1024] (bias folded in)
                const _Float16* __restrict__ Whh,  // [1024, 256] f16
                _Float16* __restrict__ encOut,     // [B*T, 640]
                int colBase, int reverse)
{
  extern __shared__ char smem[];
  _Float16* hS = (_Float16*)smem;                       // 16*256
  float*    cS = (float*)(smem + 16 * 256 * 2);         // 16*256
  float*    gS = (float*)(smem + 16 * 256 * 2 + 16 * 256 * 4); // 16*1024

  const int tid = threadIdx.x;
  const int b0  = blockIdx.x * 16;
  for (int i = tid; i < 16 * 256; i += 256) { hS[i] = (_Float16)0.0f; cS[i] = 0.0f; }
  __syncthreads();

  const int wave  = tid >> 5;
  const int lane  = tid & 31;
  const int abase = (lane < 16) ? 0 : 16;
  const int bbase = (lane < 16) ? 0 : 16;
  const int mrow  = lane & 15;
  const int rbase = (lane < 16) ? 0 : 8;

  for (int tt = 0; tt < KT; ++tt) {
    const int t = reverse ? (KT - 1 - tt) : tt;

    for (int i = 0; i < 8; ++i) {
      const int nt   = wave * 8 + i;          // 64 n-tiles of 16 -> N=1024
      const int ncol = nt * 16 + (lane & 15);
      v8f acc = (v8f){};
      for (int k0 = 0; k0 < 256; k0 += 32) {
        v8h alo = *(const v8h*)&hS[mrow * 256 + k0 + abase];
        v8h ahi = *(const v8h*)&hS[mrow * 256 + k0 + abase + 8];
        v16h af = __builtin_shufflevector(alo, ahi, 0,1,2,3,4,5,6,7,8,9,10,11,12,13,14,15);
        v16h bf = *(const v16h*)(Whh + (size_t)ncol * 256 + k0 + bbase);
        acc = __builtin_amdgcn_wmma_f32_16x16x32_f16(false, af, false, bf,
                                                     (short)0, acc, false, false);
      }
      for (int r = 0; r < 8; ++r) {
        const int row = rbase + r;
        size_t xoff = (((size_t)(b0 + row)) * KT + t) * 1024 + nt * 16 + (lane & 15);
        gS[row * 1024 + nt * 16 + (lane & 15)] = acc[r] + xw[xoff];
      }
    }
    __syncthreads();

    for (int p = tid; p < 16 * 256; p += 256) {
      const int row = p >> 8;
      const int j   = p & 255;
      const float ig = gS[row * 1024 + j];
      const float fg = gS[row * 1024 + 256 + j];
      const float gg = gS[row * 1024 + 512 + j];
      const float og = gS[row * 1024 + 768 + j];
      float cv = sigf(fg) * cS[p] + sigf(ig) * tanhf(gg);
      cS[p] = cv;
      float hv = sigf(og) * tanhf(cv);
      hS[p] = (_Float16)hv;
      encOut[(((size_t)(b0 + row)) * KT + t) * KENC_OUTP + colBase + j] = (_Float16)hv;
    }
    __syncthreads();
  }
}

// ---------------------------------------------------------------------------
// Embedding + concat -> x0 (halo layout, f16) ; also writes cond into the
// enc concat buffers (cols 512..623, zero pad 624..639).
// ---------------------------------------------------------------------------
__device__ __forceinline__ float cond_val(int j, int bt,
    const int* case_idx, const int* upos_idx, const int* lang_idx,
    const float* case_emb, const float* upos_emb, const float* lang_emb)
{
  if (j < 16) return case_emb[case_idx[bt] * 16 + j];
  if (j < 80) return upos_emb[upos_idx[bt] * 64 + (j - 16)];
  return lang_emb[lang_idx[bt] * 32 + (j - 80)];
}

__global__ void embed_kernel(const int* __restrict__ char_idx, const int* __restrict__ case_idx,
                             const int* __restrict__ upos_idx, const int* __restrict__ lang_idx,
                             const float* __restrict__ char_emb, const float* __restrict__ case_emb,
                             const float* __restrict__ upos_emb, const float* __restrict__ lang_emb,
                             _Float16* __restrict__ x0, _Float16* __restrict__ enc1,
                             _Float16* __restrict__ enc2)
{
  const int bt = blockIdx.x;                 // 0 .. B*T-1
  const int b = bt / KT, t = bt % KT;
  const int tid = threadIdx.x;
  const size_t xrow = ((size_t)b * KTH + t + 2) * KCS0P;
  if (tid < KCS0) {
    float val;
    if (tid < KCE) val = char_emb[char_idx[bt] * KCE + tid];
    else           val = cond_val(tid - KCE, bt, case_idx, upos_idx, lang_idx,
                                  case_emb, upos_emb, lang_emb);
    x0[xrow + tid] = (_Float16)val;
  }
  if (tid < 128) {
    _Float16 cv = (_Float16)0.0f;
    if (tid < KCOND)
      cv = (_Float16)cond_val(tid, bt, case_idx, upos_idx, lang_idx,
                              case_emb, upos_emb, lang_emb);
    enc1[(size_t)bt * KENC_OUTP + 512 + tid] = cv;
    enc2[(size_t)bt * KENC_OUTP + 512 + tid] = cv;
  }
}

// ---------------------------------------------------------------------------
// Conv epilogue: BN(inference) + ReLU, f32 acc -> f16 halo buffer
// ---------------------------------------------------------------------------
__global__ void bn_relu_kernel(const float* __restrict__ acc, const float* __restrict__ scale,
                               const float* __restrict__ shift, _Float16* __restrict__ xnext)
{
  const int idx = blockIdx.x * 256 + threadIdx.x;
  if (idx >= KB * KT * KCCH) return;
  const int co = idx & 511;
  const int bt = idx >> 9;
  const int b = bt / KT, t = bt % KT;
  float v = acc[idx] * scale[co] + shift[co];
  v = fmaxf(v, 0.0f);
  xnext[(((size_t)b * KTH) + t + 2) * KCCH + co] = (_Float16)v;
}

// ---------------------------------------------------------------------------
// Attention: e = v . tanh(enc_proj + q), softmax over T, ctx = att @ enc.
// One block per batch row.  Writes ctx into xin[b, 0:624].
// ---------------------------------------------------------------------------
__global__ void attn_kernel(const float* __restrict__ encproj, const float* __restrict__ q,
                            const _Float16* __restrict__ enc2, const float* __restrict__ vatt,
                            _Float16* __restrict__ xin)
{
  const int b = blockIdx.x;
  const int tid = threadIdx.x;
  __shared__ float att[KT];
  if (tid < KT) {
    const float* ep = encproj + ((size_t)b * KT + tid) * KATT;
    const float* qb = q + (size_t)b * KATT;
    float s = 0.0f;
    for (int a = 0; a < KATT; ++a) s += vatt[a] * tanhf(ep[a] + qb[a]);
    att[tid] = s;
  }
  __syncthreads();
  if (tid == 0) {
    float m = -1e30f;
    for (int t = 0; t < KT; ++t) m = fmaxf(m, att[t]);
    float sum = 0.0f;
    for (int t = 0; t < KT; ++t) { att[t] = __expf(att[t] - m); sum += att[t]; }
    float inv = 1.0f / sum;
    for (int t = 0; t < KT; ++t) att[t] *= inv;
  }
  __syncthreads();
  for (int d = tid; d < KENC_OUT; d += blockDim.x) {
    float s = 0.0f;
    for (int t = 0; t < KT; ++t)
      s += att[t] * (float)enc2[((size_t)b * KT + t) * KENC_OUTP + d];
    xin[(size_t)b * KDEC_INP + d] = (_Float16)s;
  }
}

// ---------------------------------------------------------------------------
// Decoder LSTM cell (H=512): gates [B,2048] f32 -> c f32, h f16
// ---------------------------------------------------------------------------
__global__ void dec_cell_kernel(const float* __restrict__ gates, float* __restrict__ cSt,
                                _Float16* __restrict__ hSt)
{
  const int idx = blockIdx.x * 256 + threadIdx.x;
  if (idx >= KB * KHDEC) return;
  const int b = idx >> 9, j = idx & 511;
  const float* g = gates + (size_t)b * 2048;
  const float ig = g[j], fg = g[512 + j], gg = g[1024 + j], og = g[1536 + j];
  float cv = sigf(fg) * cSt[idx] + sigf(ig) * tanhf(gg);
  cSt[idx] = cv;
  hSt[idx] = (_Float16)(sigf(og) * tanhf(cv));
}

// prev token embedding -> xin[b, 624:724]
__global__ void prev_kernel(const int* __restrict__ out_idx, const float* __restrict__ char_emb,
                            _Float16* __restrict__ xin, int scol)
{
  const int idx = blockIdx.x * 256 + threadIdx.x;
  if (idx >= KB * KCE) return;
  const int b = idx / KCE, j = idx % KCE;
  const int tok = out_idx[b * KS + scol];
  xin[(size_t)b * KDEC_INP + KENC_OUT + j] = (_Float16)char_emb[tok * KCE + j];
}

// case head: [B,3] = h1 @ Wcase^T + bcase ; one block per batch row
__global__ void out_case_kernel(const _Float16* __restrict__ h1, const float* __restrict__ W,
                                const float* __restrict__ bias, float* __restrict__ out, int s)
{
  const int b = blockIdx.x;
  const int tid = threadIdx.x;
  __shared__ float r0[256], r1[256], r2[256];
  float p0 = 0.0f, p1 = 0.0f, p2 = 0.0f;
  for (int d = tid; d < KHDEC; d += 256) {
    float hv = (float)h1[(size_t)b * KHDEC + d];
    p0 += hv * W[d];
    p1 += hv * W[512 + d];
    p2 += hv * W[1024 + d];
  }
  r0[tid] = p0; r1[tid] = p1; r2[tid] = p2;
  __syncthreads();
  for (int off = 128; off > 0; off >>= 1) {
    if (tid < off) { r0[tid] += r0[tid + off]; r1[tid] += r1[tid + off]; r2[tid] += r2[tid + off]; }
    __syncthreads();
  }
  if (tid == 0) {
    size_t base = (size_t)b * (KS * 3) + (size_t)s * 3;
    out[base + 0] = r0[0] + bias[0];
    out[base + 1] = r1[0] + bias[1];
    out[base + 2] = r2[0] + bias[2];
  }
}

// ---------------------------------------------------------------------------
// Weight / misc prep kernels
// ---------------------------------------------------------------------------
// dst[n*Kp + k] = k<K ? f16(src[n*rowStride + k*colStride + off]) : 0
__global__ void cvt_pad_kernel(const float* __restrict__ src, _Float16* __restrict__ dst,
                               int N, int K, int Kp, int rowStride, int colStride, int off)
{
  const int idx = blockIdx.x * 256 + threadIdx.x;
  if (idx >= N * Kp) return;
  const int n = idx / Kp, k = idx % Kp;
  dst[idx] = (k < K)
    ? (_Float16)src[(size_t)n * rowStride + (size_t)k * colStride + off]
    : (_Float16)0.0f;
}

__global__ void addvec_kernel(const float* a, const float* b, float* o, int n)
{
  const int i = blockIdx.x * 256 + threadIdx.x;
  if (i < n) o[i] = a[i] + b[i];
}

__global__ void conv_affine_kernel(const float* b, const float* gamma, const float* beta,
                                   const float* mean, const float* var,
                                   float* scale, float* shift)
{
  const int i = blockIdx.x * 256 + threadIdx.x;
  if (i >= KCCH) return;
  const float s = gamma[i] * rsqrtf(var[i] + 1e-5f);
  scale[i] = s;
  shift[i] = (b[i] - mean[i]) * s + beta[i];
}

__global__ void zero_f16_kernel(_Float16* p, size_t n)
{
  size_t i = (size_t)blockIdx.x * 256 + threadIdx.x;
  if (i < n) p[i] = (_Float16)0.0f;
}
__global__ void zero_f32_kernel(float* p, size_t n)
{
  size_t i = (size_t)blockIdx.x * 256 + threadIdx.x;
  if (i < n) p[i] = 0.0f;
}

// ---------------------------------------------------------------------------
// Host orchestration
// ---------------------------------------------------------------------------
extern "C" void kernel_launch(void* const* d_in, const int* in_sizes, int n_in,
                              void* d_out, int out_size, void* d_ws, size_t ws_size,
                              hipStream_t stream)
{
  (void)in_sizes; (void)n_in; (void)out_size; (void)ws_size;

  // --- inputs (setup_inputs() dict order, params in insertion order) ---
  const int*   char_idx = (const int*)d_in[0];
  const int*   case_idx = (const int*)d_in[1];
  const int*   upos_idx = (const int*)d_in[2];
  const int*   lang_idx = (const int*)d_in[3];
  const int*   out_idx  = (const int*)d_in[4];
  const float* char_emb = (const float*)d_in[5];
  const float* case_emb = (const float*)d_in[6];
  const float* upos_emb = (const float*)d_in[7];
  const float* lang_emb = (const float*)d_in[8];
  // convs[l]: W,b,gamma,beta,mean,var at 9+6l .. 14+6l
  // enc_lstm[l]: Wih_f,Whh_f,bih_f,bhh_f,Wih_b,Whh_b,bih_b,bhh_b at 27+8l
  // dec_lstm[l]: Wih,Whh,bih,bhh at 43+4l
  const float* att_We = (const float*)d_in[51];
  const float* att_be = (const float*)d_in[52];
  const float* att_Wd = (const float*)d_in[53];
  const float* att_v  = (const float*)d_in[54];
  const float* ocW    = (const float*)d_in[55];
  const float* ocB    = (const float*)d_in[56];
  const float* ocaW   = (const float*)d_in[57];
  const float* ocaB   = (const float*)d_in[58];

  float* out_char = (float*)d_out;                             // [B,S,512]
  float* out_case = out_char + (size_t)KB * KS * KVV;          // [B,S,3]

  // --- workspace bump allocator ---
  char* wsp = (char*)d_ws;
  size_t wo = 0;
  auto alloc = [&](size_t bytes) -> char* {
    char* r = wsp + wo;
    wo = (wo + bytes + 255) & ~(size_t)255;
    return r;
  };

  _Float16* x0h   = (_Float16*)alloc((size_t)KB * KTH * KCS0P * 2);
  _Float16* xA    = (_Float16*)alloc((size_t)KB * KTH * KCCH * 2);
  _Float16* xB    = (_Float16*)alloc((size_t)KB * KTH * KCCH * 2);
  float*    accxw = (float*)   alloc((size_t)KB * KT * 1024 * 4);   // conv acc / xw
  _Float16* enc1  = (_Float16*)alloc((size_t)KB * KT * KENC_OUTP * 2);
  _Float16* enc2  = (_Float16*)alloc((size_t)KB * KT * KENC_OUTP * 2);
  float*    eproj = (float*)   alloc((size_t)KB * KT * KATT * 4);
  float*    qbuf  = (float*)   alloc((size_t)KB * KATT * 4);
  float*    gates = (float*)   alloc((size_t)KB * 2048 * 4);
  _Float16* xin   = (_Float16*)alloc((size_t)KB * KDEC_INP * 2);
  _Float16* h0    = (_Float16*)alloc((size_t)KB * KHDEC * 2);
  _Float16* h1    = (_Float16*)alloc((size_t)KB * KHDEC * 2);
  float*    c0    = (float*)   alloc((size_t)KB * KHDEC * 4);
  float*    c1    = (float*)   alloc((size_t)KB * KHDEC * 4);

  // f16 weights
  _Float16* convW16[3][5];
  float *convScale[3], *convShift[3];
  for (int l = 0; l < 3; ++l) {
    int cinp = (l == 0) ? KCS0P : KCCH;
    for (int tap = 0; tap < 5; ++tap)
      convW16[l][tap] = (_Float16*)alloc((size_t)KCCH * cinp * 2);
    convScale[l] = (float*)alloc(KCCH * 4);
    convShift[l] = (float*)alloc(KCCH * 4);
  }
  _Float16 *encWih[2][2], *encWhh[2][2];
  float* encB[2][2];
  for (int l = 0; l < 2; ++l)
    for (int d = 0; d < 2; ++d) {
      int kin = (l == 0) ? KCCH : KENC_OUTP;
      encWih[l][d] = (_Float16*)alloc((size_t)1024 * kin * 2);
      encWhh[l][d] = (_Float16*)alloc((size_t)1024 * 256 * 2);
      encB[l][d]   = (float*)alloc(1024 * 4);
    }
  _Float16* decWih0 = (_Float16*)alloc((size_t)2048 * KDEC_INP * 2);
  _Float16* decWhh0 = (_Float16*)alloc((size_t)2048 * KHDEC * 2);
  _Float16* decWih1 = (_Float16*)alloc((size_t)2048 * KHDEC * 2);
  _Float16* decWhh1 = (_Float16*)alloc((size_t)2048 * KHDEC * 2);
  float* decB0 = (float*)alloc(2048 * 4);
  float* decB1 = (float*)alloc(2048 * 4);
  _Float16* We16 = (_Float16*)alloc((size_t)KATT * KENC_OUTP * 2);
  _Float16* Wd16 = (_Float16*)alloc((size_t)KATT * KHDEC * 2);
  _Float16* Wc16 = (_Float16*)alloc((size_t)KVV * KHDEC * 2);

  auto nb = [](size_t n) { return (unsigned)((n + 255) / 256); };

  auto gemm = [&](const _Float16* A, const _Float16* Bw, float* C, const float* bias,
                  int M, int N, int Kp, int lda, int ldc, int accum,
                  int rps, int sstr, int roff) {
    dim3 g((unsigned)(N / 64), (unsigned)(M / 64));
    gemm_kernel<<<g, dim3(256), 0, stream>>>(A, Bw, C, bias, Kp, lda, ldc, accum,
                                             rps, sstr, roff);
  };

  // ============================ weight prep ============================
  for (int l = 0; l < 3; ++l) {
    int cin  = (l == 0) ? KCS0 : KCCH;
    int cinp = (l == 0) ? KCS0P : KCCH;
    const float* W    = (const float*)d_in[9 + 6 * l];
    const float* bb   = (const float*)d_in[10 + 6 * l];
    const float* gam  = (const float*)d_in[11 + 6 * l];
    const float* bet  = (const float*)d_in[12 + 6 * l];
    const float* mn   = (const float*)d_in[13 + 6 * l];
    const float* vr   = (const float*)d_in[14 + 6 * l];
    for (int tap = 0; tap < 5; ++tap)
      cvt_pad_kernel<<<nb((size_t)KCCH * cinp), 256, 0, stream>>>(
          W, convW16[l][tap], KCCH, cin, cinp, cin * 5, 5, tap);
    conv_affine_kernel<<<2, 256, 0, stream>>>(bb, gam, bet, mn, vr,
                                              convScale[l], convShift[l]);
  }
  for (int l = 0; l < 2; ++l) {
    int ksrc = (l == 0) ? KCCH : KENC_OUT;
    int kpad = (l == 0) ? KCCH : KENC_OUTP;
    for (int d = 0; d < 2; ++d) {
      int base = 27 + 8 * l + 4 * d;
      cvt_pad_kernel<<<nb((size_t)1024 * kpad), 256, 0, stream>>>(
          (const float*)d_in[base + 0], encWih[l][d], 1024, ksrc, kpad, ksrc, 1, 0);
      cvt_pad_kernel<<<nb((size_t)1024 * 256), 256, 0, stream>>>(
          (const float*)d_in[base + 1], encWhh[l][d], 1024, 256, 256, 256, 1, 0);
      addvec_kernel<<<4, 256, 0, stream>>>((const float*)d_in[base + 2],
                                           (const float*)d_in[base + 3], encB[l][d], 1024);
    }
  }
  cvt_pad_kernel<<<nb((size_t)2048 * KDEC_INP), 256, 0, stream>>>(
      (const float*)d_in[43], decWih0, 2048, KDEC_IN, KDEC_INP, KDEC_IN, 1, 0);
  cvt_pad_kernel<<<nb((size_t)2048 * KHDEC), 256, 0, stream>>>(
      (const float*)d_in[44], decWhh0, 2048, KHDEC, KHDEC, KHDEC, 1, 0);
  addvec_kernel<<<8, 256, 0, stream>>>((const float*)d_in[45], (const float*)d_in[46],
                                       decB0, 2048);
  cvt_pad_kernel<<<nb((size_t)2048 * KHDEC), 256, 0, stream>>>(
      (const float*)d_in[47], decWih1, 2048, KHDEC, KHDEC, KHDEC, 1, 0);
  cvt_pad_kernel<<<nb((size_t)2048 * KHDEC), 256, 0, stream>>>(
      (const float*)d_in[48], decWhh1, 2048, KHDEC, KHDEC, KHDEC, 1, 0);
  addvec_kernel<<<8, 256, 0, stream>>>((const float*)d_in[49], (const float*)d_in[50],
                                       decB1, 2048);
  cvt_pad_kernel<<<nb((size_t)KATT * KENC_OUTP), 256, 0, stream>>>(
      att_We, We16, KATT, KENC_OUT, KENC_OUTP, KENC_OUT, 1, 0);
  cvt_pad_kernel<<<nb((size_t)KATT * KHDEC), 256, 0, stream>>>(
      att_Wd, Wd16, KATT, KHDEC, KHDEC, KHDEC, 1, 0);
  cvt_pad_kernel<<<nb((size_t)KVV * KHDEC), 256, 0, stream>>>(
      ocW, Wc16, KVV, KHDEC, KHDEC, KHDEC, 1, 0);

  // ============================ activations init ============================
  zero_f16_kernel<<<nb((size_t)KB * KTH * KCS0P), 256, 0, stream>>>(x0h, (size_t)KB * KTH * KCS0P);
  zero_f16_kernel<<<nb((size_t)KB * KTH * KCCH), 256, 0, stream>>>(xA, (size_t)KB * KTH * KCCH);
  zero_f16_kernel<<<nb((size_t)KB * KTH * KCCH), 256, 0, stream>>>(xB, (size_t)KB * KTH * KCCH);
  zero_f16_kernel<<<nb((size_t)KB * KDEC_INP), 256, 0, stream>>>(xin, (size_t)KB * KDEC_INP);
  zero_f16_kernel<<<nb((size_t)KB * KHDEC), 256, 0, stream>>>(h0, (size_t)KB * KHDEC);
  zero_f16_kernel<<<nb((size_t)KB * KHDEC), 256, 0, stream>>>(h1, (size_t)KB * KHDEC);
  zero_f32_kernel<<<nb((size_t)KB * KHDEC), 256, 0, stream>>>(c0, (size_t)KB * KHDEC);
  zero_f32_kernel<<<nb((size_t)KB * KHDEC), 256, 0, stream>>>(c1, (size_t)KB * KHDEC);

  embed_kernel<<<KB * KT, 256, 0, stream>>>(char_idx, case_idx, upos_idx, lang_idx,
                                            char_emb, case_emb, upos_emb, lang_emb,
                                            x0h, enc1, enc2);

  // ============================ conv stack ============================
  const int M = KB * KT;  // 49152
  for (int tap = 0; tap < 5; ++tap)  // conv1: x0h -> acc
    gemm(x0h, convW16[0][tap], accxw, nullptr, M, KCCH, KCS0P, KCS0P, KCCH,
         tap > 0, KT, KTH, tap);
  bn_relu_kernel<<<nb((size_t)M * KCCH), 256, 0, stream>>>(accxw, convScale[0], convShift[0], xA);
  for (int tap = 0; tap < 5; ++tap)  // conv2: xA -> acc
    gemm(xA, convW16[1][tap], accxw, nullptr, M, KCCH, KCCH, KCCH, KCCH,
         tap > 0, KT, KTH, tap);
  bn_relu_kernel<<<nb((size_t)M * KCCH), 256, 0, stream>>>(accxw, convScale[1], convShift[1], xB);
  for (int tap = 0; tap < 5; ++tap)  // conv3: xB -> acc
    gemm(xB, convW16[2][tap], accxw, nullptr, M, KCCH, KCCH, KCCH, KCCH,
         tap > 0, KT, KTH, tap);
  bn_relu_kernel<<<nb((size_t)M * KCCH), 256, 0, stream>>>(accxw, convScale[2], convShift[2], xA);

  // ============================ BiLSTM encoder ============================
  const size_t scanLds = 16 * 256 * 2 + 16 * 256 * 4 + 16 * 1024 * 4;  // 88KB
  // layer 0 (input = conv output xA, halo layout)
  for (int d = 0; d < 2; ++d) {
    gemm(xA, encWih[0][d], accxw, encB[0][d], M, 1024, KCCH, KCCH, 1024, 0,
         KT, KTH, 2);
    enc_scan_kernel<<<KB / 16, 256, scanLds, stream>>>(accxw, encWhh[0][d], enc1,
                                                       d == 0 ? 0 : 256, d);
  }
  // layer 1 (input = enc1, dense)
  for (int d = 0; d < 2; ++d) {
    gemm(enc1, encWih[1][d], accxw, encB[1][d], M, 1024, KENC_OUTP, KENC_OUTP,
         1024, 0, 0, 0, 0);
    enc_scan_kernel<<<KB / 16, 256, scanLds, stream>>>(accxw, encWhh[1][d], enc2,
                                                       d == 0 ? 0 : 256, d);
  }
  // enc_proj = enc2 @ We^T + be
  gemm(enc2, We16, eproj, att_be, M, KATT, KENC_OUTP, KENC_OUTP, KATT, 0, 0, 0, 0);

  // ============================ attention decoder ============================
  for (int s = 0; s < KS; ++s) {
    // q = h1 @ Wd^T
    gemm(h1, Wd16, qbuf, nullptr, KB, KATT, KHDEC, KHDEC, KATT, 0, 0, 0, 0);
    // attention -> ctx into xin[0:624]
    attn_kernel<<<KB, 256, 0, stream>>>(eproj, qbuf, enc2, att_v, xin);
    // prev-token embedding into xin[624:724] (zeros at s==0)
    if (s > 0)
      prev_kernel<<<nb((size_t)KB * KCE), 256, 0, stream>>>(out_idx, char_emb, xin, s - 1);
    // decoder layer 0
    gemm(xin, decWih0, gates, decB0, KB, 2048, KDEC_INP, KDEC_INP, 2048, 0, 0, 0, 0);
    gemm(h0, decWhh0, gates, nullptr, KB, 2048, KHDEC, KHDEC, 2048, 1, 0, 0, 0);
    dec_cell_kernel<<<nb((size_t)KB * KHDEC), 256, 0, stream>>>(gates, c0, h0);
    // decoder layer 1 (reads old h1 before the cell overwrites it)
    gemm(h0, decWih1, gates, decB1, KB, 2048, KHDEC, KHDEC, 2048, 0, 0, 0, 0);
    gemm(h1, decWhh1, gates, nullptr, KB, 2048, KHDEC, KHDEC, 2048, 1, 0, 0, 0);
    dec_cell_kernel<<<nb((size_t)KB * KHDEC), 256, 0, stream>>>(gates, c1, h1);
    // char logits straight into d_out (row stride S*V, column offset s*V)
    gemm(h1, Wc16, out_char + (size_t)s * KVV, ocB, KB, KVV, KHDEC, KHDEC,
         KS * KVV, 0, 0, 0, 0);
    // case logits
    out_case_kernel<<<KB, 256, 0, stream>>>(h1, ocaW, ocaB, out_case, s);
  }
}
